// VisionTransformer_1194000908903
// MI455X (gfx1250) — compile-verified, matrix operand order checked by reading, not run on hardware
//
#include <hip/hip_runtime.h>
#include <math.h>

// ---------------------------------------------------------------------------
// Types / helpers
// ---------------------------------------------------------------------------
typedef __attribute__((ext_vector_type(16))) __bf16          v16bf;
typedef __attribute__((ext_vector_type(16))) unsigned short  v16us;
typedef __attribute__((ext_vector_type(8)))  unsigned short  v8us;
typedef __attribute__((ext_vector_type(8)))  float           v8f;

#define DEV static __device__ __forceinline__

DEV unsigned short f2bf(float f) {
  union { float f; unsigned int u; } v; v.f = f;
  unsigned int r = v.u + 0x7fffu + ((v.u >> 16) & 1u);   // round-to-nearest-even
  return (unsigned short)(r >> 16);
}
DEV float bf2f(unsigned short h) {
  union { unsigned int u; float f; } v; v.u = ((unsigned int)h) << 16; return v.f;
}
DEV float gelu_exact(float x) { return 0.5f * x * (1.0f + erff(x * 0.70710678118654752f)); }

// Model constants
#define BATCH 32
#define EMB   768
#define NTOK  197
#define NHEAD 12
#define HDIM  64
#define DFF   3072
#define DEPTH 12
#define MROWS (BATCH * NTOK)      // 6304 (multiple of 16)
#define NPAD  208                 // token rows padded for WMMA M tiles
#define KPAD  224                 // token dim padded for P@V K-loop (7*32)
#define BH    (BATCH * NHEAD)     // 384

// ---------------------------------------------------------------------------
// Fragment loader: 16-bit WMMA operand layout.
// Element i of lane (hk = lane>>4):  k = (i&8)*2 + hk*8 + (i&7)
//  -> two contiguous 8-ushort (16B) loads at offsets {0, +16} from
//     (row_base + kk + hk*8). Caller guarantees in-bounds (rows are clamped).
// ---------------------------------------------------------------------------
DEV v16us ld_frag(const unsigned short* __restrict__ p) {
  v8us lo = *(const v8us*)p;          // k .. k+7
  v8us hi = *(const v8us*)(p + 16);   // k+16 .. k+23
  v16us f;
  #pragma unroll
  for (int i = 0; i < 8; ++i) { f[i] = lo[i]; f[8 + i] = hi[i]; }
  return f;
}

// ---------------------------------------------------------------------------
// WMMA GEMM:  C = act(scale * A(bf16)[M,K] x B(bf16)[K,N] + bias) + resid
//   - A row-major [M,K] (lda); B stored COLUMN-major, i.e. Bt[N,K] (ldb)
//   - z batches: A += z*sA, Bt += z*sB, C += (z/czdiv)*sCb + (z%czdiv)*sCh
//   - output fp32 (Cf) or bf16 (Cb)
// Block: 256 thr = 8 waves (4 in M x 2 in N); wave -> 32x32 tile (4 WMMAs/step).
// Block tile: 128(M) x 64(N). No LDS, no barriers, no exec masking in loop:
// out-of-range rows are CLAMPED (their products land only in C rows/cols that
// the store epilogue discards), so all loads are unconditional. Fragments are
// double-buffered (next-step loads issue before this step's WMMAs) and the
// final K-step is peeled so loop offsets stay affine (immediate offsets, no
// per-iteration 64-bit address rebuilds).
// ---------------------------------------------------------------------------
__global__ __launch_bounds__(256) void gemm_bf16_wmma(
    const unsigned short* __restrict__ A, long long sA, int lda,
    const unsigned short* __restrict__ Bt, long long sB, int ldb,
    const float* __restrict__ bias,
    const float* __restrict__ resid, long long sR, int ldr,
    float* __restrict__ Cf, unsigned short* __restrict__ Cb,
    int czdiv, long long sCb_, long long sCh_, int ldc,
    int Mv, int Nv, int K, float scale, int act)
{
  const int z   = blockIdx.z;
  const unsigned short* Az = A  + (size_t)z * (size_t)sA;
  const unsigned short* Bz = Bt + (size_t)z * (size_t)sB;
  const long long coff = (long long)(z / czdiv) * sCb_ + (long long)(z % czdiv) * sCh_;

  const int t    = threadIdx.x;
  const int wv   = t >> 5;
  const int lane = t & 31;
  const int hk   = lane >> 4;
  const int l15  = lane & 15;

  const int tm = blockIdx.y * 128 + (wv >> 1) * 32;
  const int tn = blockIdx.x * 64  + (wv & 1) * 32;

  const int am0 = tm + l15,  am1 = am0 + 16;
  const int bn0 = tn + l15,  bn1 = bn0 + 16;

  // Clamp rows for loading; correctness preserved because clamped rows only
  // contribute to outputs masked off in the epilogue.
  const int amc0 = (am0 < Mv) ? am0 : (Mv - 1);
  const int amc1 = (am1 < Mv) ? am1 : (Mv - 1);
  const int bnc0 = (bn0 < Nv) ? bn0 : (Nv - 1);
  const int bnc1 = (bn1 < Nv) ? bn1 : (Nv - 1);

  const unsigned short* pa0 = Az + (size_t)amc0 * (size_t)lda + (hk << 3);
  const unsigned short* pa1 = Az + (size_t)amc1 * (size_t)lda + (hk << 3);
  const unsigned short* pb0 = Bz + (size_t)bnc0 * (size_t)ldb + (hk << 3);
  const unsigned short* pb1 = Bz + (size_t)bnc1 * (size_t)ldb + (hk << 3);

  v8f acc00 = {}, acc01 = {}, acc10 = {}, acc11 = {};

  // Prologue loads (kk = 0)
  v16us a0 = ld_frag(pa0), a1 = ld_frag(pa1);
  v16us b0 = ld_frag(pb0), b1 = ld_frag(pb1);

  // Main loop: body handles step (kk-32), loads step kk. Final step peeled.
  for (int kk = 32; kk < K; kk += 32) {
    __builtin_prefetch(pa0 + kk + 32, 0, 0);
    __builtin_prefetch(pb0 + kk + 32, 0, 0);

    // Issue next-step loads before this step's WMMAs.
    const v16us na0 = ld_frag(pa0 + kk);
    const v16us na1 = ld_frag(pa1 + kk);
    const v16us nb0 = ld_frag(pb0 + kk);
    const v16us nb1 = ld_frag(pb1 + kk);

    acc00 = __builtin_amdgcn_wmma_f32_16x16x32_bf16(false, __builtin_bit_cast(v16bf, a0),
                                                    false, __builtin_bit_cast(v16bf, b0),
                                                    (short)0, acc00, false, false);
    acc01 = __builtin_amdgcn_wmma_f32_16x16x32_bf16(false, __builtin_bit_cast(v16bf, a0),
                                                    false, __builtin_bit_cast(v16bf, b1),
                                                    (short)0, acc01, false, false);
    acc10 = __builtin_amdgcn_wmma_f32_16x16x32_bf16(false, __builtin_bit_cast(v16bf, a1),
                                                    false, __builtin_bit_cast(v16bf, b0),
                                                    (short)0, acc10, false, false);
    acc11 = __builtin_amdgcn_wmma_f32_16x16x32_bf16(false, __builtin_bit_cast(v16bf, a1),
                                                    false, __builtin_bit_cast(v16bf, b1),
                                                    (short)0, acc11, false, false);

    a0 = na0; a1 = na1; b0 = nb0; b1 = nb1;
  }

  // Peeled final step: no trailing loads.
  acc00 = __builtin_amdgcn_wmma_f32_16x16x32_bf16(false, __builtin_bit_cast(v16bf, a0),
                                                  false, __builtin_bit_cast(v16bf, b0),
                                                  (short)0, acc00, false, false);
  acc01 = __builtin_amdgcn_wmma_f32_16x16x32_bf16(false, __builtin_bit_cast(v16bf, a0),
                                                  false, __builtin_bit_cast(v16bf, b1),
                                                  (short)0, acc01, false, false);
  acc10 = __builtin_amdgcn_wmma_f32_16x16x32_bf16(false, __builtin_bit_cast(v16bf, a1),
                                                  false, __builtin_bit_cast(v16bf, b0),
                                                  (short)0, acc10, false, false);
  acc11 = __builtin_amdgcn_wmma_f32_16x16x32_bf16(false, __builtin_bit_cast(v16bf, a1),
                                                  false, __builtin_bit_cast(v16bf, b1),
                                                  (short)0, acc11, false, false);

  // ---- store 4 sub-tiles ----
  v8f accs[2][2];
  accs[0][0] = acc00; accs[0][1] = acc01; accs[1][0] = acc10; accs[1][1] = acc11;
  #pragma unroll
  for (int tj = 0; tj < 2; ++tj) {
    const int n = tn + tj * 16 + l15;
    if (n < Nv) {
      const float bvn = bias ? bias[n] : 0.0f;
      #pragma unroll
      for (int ti = 0; ti < 2; ++ti) {
        const v8f acc = accs[ti][tj];
        #pragma unroll
        for (int e = 0; e < 8; ++e) {
          const int m = tm + ti * 16 + (hk << 3) + e;
          if (m < Mv) {
            float v = acc[e] * scale + bvn;
            if (act == 1) v = gelu_exact(v);
            if (resid) v += resid[(size_t)z * (size_t)sR + (size_t)m * (size_t)ldr + n];
            const size_t o = (size_t)coff + (size_t)m * (size_t)ldc + n;
            if (Cf) Cf[o] = v;
            else    Cb[o] = f2bf(v);
          }
        }
      }
    }
  }
}

// ---------------------------------------------------------------------------
// Patch embedding + cls token + pos embed -> x0 fp32 [B,197,768]
// ---------------------------------------------------------------------------
__global__ __launch_bounds__(256) void patch_embed_kernel(
    const float* __restrict__ x, const float* __restrict__ pw,
    const float* __restrict__ pb, const float* __restrict__ cls,
    const float* __restrict__ pos, float* __restrict__ xout)
{
  const size_t idx = (size_t)blockIdx.x * 256 + threadIdx.x;
  const size_t total = (size_t)BATCH * NTOK * EMB;
  if (idx >= total) return;
  const int e = (int)(idx % EMB);
  const int tk = (int)((idx / EMB) % NTOK);
  const int b  = (int)(idx / ((size_t)EMB * NTOK));
  float v;
  if (tk == 0) {
    v = cls[e] + pos[e];
  } else {
    const int tp = tk - 1, gh = tp / 14, gw = tp % 14;
    const float* xb = x + (size_t)b * 3 * 224 * 224;
    const float* w  = pw + (size_t)e * 768;   // [c][16][16]
    float s = 0.0f;
    for (int c = 0; c < 3; ++c) {
      const float* xc = xb + (size_t)c * 224 * 224 + (size_t)(gh * 16) * 224 + gw * 16;
      const float* wc = w + c * 256;
      #pragma unroll 4
      for (int py = 0; py < 16; ++py)
        for (int px = 0; px < 16; ++px)
          s += xc[py * 224 + px] * wc[py * 16 + px];
    }
    v = s + pb[e] + pos[(size_t)tk * EMB + e];
  }
  xout[idx] = v;
}

// ---------------------------------------------------------------------------
// LayerNorm over width 768; out bf16 (outb) or fp32 (outf)
// ---------------------------------------------------------------------------
__global__ __launch_bounds__(256) void ln_rows(
    const float* __restrict__ xin, long long rs_in,
    const float* __restrict__ gs, const float* __restrict__ gb,
    unsigned short* __restrict__ outb, float* __restrict__ outf,
    long long rs_out)
{
  __shared__ float red[256];
  const float* xr = xin + (size_t)blockIdx.x * (size_t)rs_in;
  const int t = threadIdx.x;
  const float v0 = xr[t], v1 = xr[t + 256], v2 = xr[t + 512];
  red[t] = v0 + v1 + v2;
  __syncthreads();
  for (int o = 128; o > 0; o >>= 1) { if (t < o) red[t] += red[t + o]; __syncthreads(); }
  const float mu = red[0] * (1.0f / 768.0f);
  __syncthreads();
  const float d0 = v0 - mu, d1 = v1 - mu, d2 = v2 - mu;
  red[t] = d0 * d0 + d1 * d1 + d2 * d2;
  __syncthreads();
  for (int o = 128; o > 0; o >>= 1) { if (t < o) red[t] += red[t + o]; __syncthreads(); }
  const float inv = rsqrtf(red[0] * (1.0f / 768.0f) + 1e-6f);
  const float y0 = d0 * inv * gs[t]       + gb[t];
  const float y1 = d1 * inv * gs[t + 256] + gb[t + 256];
  const float y2 = d2 * inv * gs[t + 512] + gb[t + 512];
  const size_t o0 = (size_t)blockIdx.x * (size_t)rs_out;
  if (outb) { outb[o0 + t] = f2bf(y0); outb[o0 + t + 256] = f2bf(y1); outb[o0 + t + 512] = f2bf(y2); }
  else      { outf[o0 + t] = y0;       outf[o0 + t + 256] = y1;       outf[o0 + t + 512] = y2; }
}

// ---------------------------------------------------------------------------
// qkv bf16 [6304,2304] -> Qb,Kb [BH,208,64] (row=token), Vt [BH,64,224]
// (V stored transposed: row=d, col=token; all pads zero-filled)
// ---------------------------------------------------------------------------
__global__ __launch_bounds__(256) void split_qkv(
    const unsigned short* __restrict__ qkv,
    unsigned short* __restrict__ Qb, unsigned short* __restrict__ Kb,
    unsigned short* __restrict__ Vt)
{
  const size_t idx = (size_t)blockIdx.x * 256 + threadIdx.x;
  const size_t total = (size_t)BH * KPAD * HDIM;
  if (idx >= total) return;
  const int d = (int)(idx & 63);
  const size_t r = idx >> 6;                // bh*224 + m
  const int m  = (int)(r % KPAD);
  const size_t bh = r / KPAD;
  const int h = (int)(bh % NHEAD);
  const int b = (int)(bh / NHEAD);
  unsigned short q = 0, k = 0, v = 0;
  if (m < NTOK) {
    const unsigned short* row = qkv + (size_t)(b * NTOK + m) * (3 * EMB) + h * HDIM + d;
    q = row[0]; k = row[EMB]; v = row[2 * EMB];
  }
  if (m < NPAD) {
    const size_t o = (bh * NPAD + m) * HDIM + d;
    Qb[o] = q; Kb[o] = k;
  }
  Vt[(bh * HDIM + d) * KPAD + m] = v;
}

// ---------------------------------------------------------------------------
// Row softmax: S fp32 [BH,208,224] (197 valid cols) -> P bf16 (cols>=197 zero)
// ---------------------------------------------------------------------------
__global__ __launch_bounds__(224) void softmax_rows(
    const float* __restrict__ S, unsigned short* __restrict__ P)
{
  __shared__ float red[224];
  const int r = blockIdx.x;            // 0 .. BH*197-1
  const int m = r % NTOK;
  const int bh = r / NTOK;
  const float* sr = S + ((size_t)bh * NPAD + m) * KPAD;
  unsigned short* pr = P + ((size_t)bh * NPAD + m) * KPAD;
  const int t = threadIdx.x;
  const float v = (t < NTOK) ? sr[t] : -3.0e38f;
  red[t] = v;
  __syncthreads();
  for (int o = 128; o > 0; o >>= 1) {
    if (t < o && t + o < 224) red[t] = fmaxf(red[t], red[t + o]);
    __syncthreads();
  }
  const float mx = red[0];
  __syncthreads();
  const float ex = (t < NTOK) ? __expf(v - mx) : 0.0f;
  red[t] = ex;
  __syncthreads();
  for (int o = 128; o > 0; o >>= 1) {
    if (t < o && t + o < 224) red[t] += red[t + o];
    __syncthreads();
  }
  const float inv = 1.0f / red[0];
  pr[t] = (t < NTOK) ? f2bf(ex * inv) : (unsigned short)0;
}

// ---------------------------------------------------------------------------
// Attention statistics (last block)
// ---------------------------------------------------------------------------
__global__ __launch_bounds__(256) void attn_back_mean(
    const unsigned short* __restrict__ P, float* __restrict__ out)
{
  const int idx = blockIdx.x * 256 + threadIdx.x;
  if (idx >= BATCH * NTOK * NTOK) return;
  const int m = idx % NTOK;
  const int n = (idx / NTOK) % NTOK;
  const int b = idx / (NTOK * NTOK);
  float s = 0.0f;
  #pragma unroll
  for (int h = 0; h < NHEAD; ++h)
    s += bf2f(P[(((size_t)(b * NHEAD + h)) * NPAD + n) * KPAD + m]);
  out[idx] = s * (1.0f / 12.0f);
}

__global__ __launch_bounds__(256) void cls_attn_mean(
    const unsigned short* __restrict__ P, float* __restrict__ ca)
{
  const int idx = blockIdx.x * 256 + threadIdx.x;
  if (idx >= BATCH * NTOK) return;
  const int j = idx % NTOK;
  const int b = idx / NTOK;
  float s = 0.0f;
  #pragma unroll
  for (int h = 0; h < NHEAD; ++h)
    s += bf2f(P[(((size_t)(b * NHEAD + h)) * NPAD + 0) * KPAD + j]);
  ca[idx] = s * (1.0f / 12.0f);
}

__global__ __launch_bounds__(256) void cls_patch_kernel(
    const float* __restrict__ ab, const float* __restrict__ ca,
    float* __restrict__ out)
{
  const int idx = blockIdx.x * 256 + threadIdx.x;
  if (idx >= BATCH * 196) return;
  const int j = idx % 196;
  const int b = idx / 196;
  const float* abb = ab + (size_t)b * NTOK * NTOK;
  const float* cab = ca + b * NTOK;
  float s = 0.0f;
  for (int n = 0; n < NTOK; ++n) s += abb[(size_t)n * NTOK + (j + 1)] * cab[n];
  out[idx] = s * (1.0f / 197.0f);
}

// ---------------------------------------------------------------------------
// fp32 [K,N] -> bf16 transposed [N,K]  (coalesced writes)
// ---------------------------------------------------------------------------
__global__ __launch_bounds__(256) void cvt_f32_bf16_t(
    const float* __restrict__ in, unsigned short* __restrict__ out, int K, int N)
{
  const int j = blockIdx.x * 256 + threadIdx.x;
  if (j < K * N) {
    const int n = j / K;
    const int k = j % K;
    out[j] = f2bf(in[(size_t)k * N + n]);
  }
}

// ---------------------------------------------------------------------------
// Host launcher
// ---------------------------------------------------------------------------
static inline int cdiv_h(int a, int b) { return (a + b - 1) / b; }

extern "C" void kernel_launch(void* const* d_in, const int* in_sizes, int n_in,
                              void* d_out, int out_size, void* d_ws, size_t ws_size,
                              hipStream_t stream) {
  (void)in_sizes; (void)n_in; (void)out_size; (void)ws_size;

  const float* x_in    = (const float*)d_in[0];
  const float* patch_w = (const float*)d_in[1];
  const float* patch_b = (const float*)d_in[2];
  const float* cls_tok = (const float*)d_in[3];
  const float* pos_emb = (const float*)d_in[4];
  const float* ln1_s   = (const float*)d_in[5];
  const float* ln1_b   = (const float*)d_in[6];
  const float* qkv_w   = (const float*)d_in[7];
  const float* qkv_b   = (const float*)d_in[8];
  const float* proj_w  = (const float*)d_in[9];
  const float* proj_b  = (const float*)d_in[10];
  const float* ln2_s   = (const float*)d_in[11];
  const float* ln2_b   = (const float*)d_in[12];
  const float* fc1_w   = (const float*)d_in[13];
  const float* fc1_b   = (const float*)d_in[14];
  const float* fc2_w   = (const float*)d_in[15];
  const float* fc2_b   = (const float*)d_in[16];
  const float* lnf_s   = (const float*)d_in[17];
  const float* lnf_b   = (const float*)d_in[18];

  // ---- workspace carve (256B aligned) ----
  char* p = (char*)d_ws;
  auto carve = [&](size_t bytes) -> void* {
    void* r = (void*)p;
    p += (bytes + 255) & ~(size_t)255;
    return r;
  };
  float*          xA   = (float*)carve((size_t)MROWS * EMB * 4);
  float*          xB   = (float*)carve((size_t)MROWS * EMB * 4);
  unsigned short* ybuf = (unsigned short*)carve((size_t)MROWS * EMB * 2);
  unsigned short* qkvb = (unsigned short*)carve((size_t)MROWS * 3 * EMB * 2);
  unsigned short* Qb   = (unsigned short*)carve((size_t)BH * NPAD * HDIM * 2);
  unsigned short* Kb   = (unsigned short*)carve((size_t)BH * NPAD * HDIM * 2);
  unsigned short* Vt   = (unsigned short*)carve((size_t)BH * HDIM * KPAD * 2);
  float*          Sb   = (float*)carve((size_t)BH * NPAD * KPAD * 4);
  unsigned short* Pb   = (unsigned short*)carve((size_t)BH * NPAD * KPAD * 2);
  unsigned short* ob   = (unsigned short*)carve((size_t)MROWS * EMB * 2);
  unsigned short* h1   = (unsigned short*)carve((size_t)MROWS * DFF * 2);
  unsigned short* wq   = (unsigned short*)carve((size_t)EMB * 3 * EMB * 2);  // [2304,768]
  unsigned short* wp   = (unsigned short*)carve((size_t)EMB * EMB * 2);      // [768,768]
  unsigned short* w1   = (unsigned short*)carve((size_t)EMB * DFF * 2);      // [3072,768]
  unsigned short* w2   = (unsigned short*)carve((size_t)DFF * EMB * 2);      // [768,3072]
  float*          cab  = (float*)carve((size_t)BATCH * NTOK * 4);

  float* out_cls = (float*)d_out;                        // [32,768]
  float* out_ab  = out_cls + (size_t)BATCH * EMB;        // [32,197,197]
  float* out_cp  = out_ab + (size_t)BATCH * NTOK * NTOK; // [32,196]

  // ---- patch embedding ----
  {
    const size_t total = (size_t)BATCH * NTOK * EMB;
    patch_embed_kernel<<<(unsigned)((total + 255) / 256), 256, 0, stream>>>(
        x_in, patch_w, patch_b, cls_tok, pos_emb, xA);
  }

  const float attn_scale = 0.125f;   // 64^-0.5

  for (int l = 0; l < DEPTH; ++l) {
    // ---- convert + transpose this block's weights to bf16 [N,K] ----
    const int nwq = EMB * 3 * EMB, nwp = EMB * EMB, nw1 = EMB * DFF, nw2 = DFF * EMB;
    cvt_f32_bf16_t<<<cdiv_h(nwq, 256), 256, 0, stream>>>(qkv_w + (size_t)l * nwq, wq, EMB, 3 * EMB);
    cvt_f32_bf16_t<<<cdiv_h(nwp, 256), 256, 0, stream>>>(proj_w + (size_t)l * nwp, wp, EMB, EMB);
    cvt_f32_bf16_t<<<cdiv_h(nw1, 256), 256, 0, stream>>>(fc1_w + (size_t)l * nw1, w1, EMB, DFF);
    cvt_f32_bf16_t<<<cdiv_h(nw2, 256), 256, 0, stream>>>(fc2_w + (size_t)l * nw2, w2, DFF, EMB);

    // ---- LN1 -> ybuf (bf16) ----
    ln_rows<<<MROWS, 256, 0, stream>>>(xA, EMB, ln1_s + (size_t)l * EMB, ln1_b + (size_t)l * EMB,
                                       ybuf, nullptr, EMB);

    // ---- qkv = ybuf @ Wqkv + b  -> qkvb (bf16) ----
    gemm_bf16_wmma<<<dim3(cdiv_h(3 * EMB, 64), cdiv_h(MROWS, 128), 1), 256, 0, stream>>>(
        ybuf, 0, EMB, wq, 0, EMB,
        qkv_b + (size_t)l * 3 * EMB,
        nullptr, 0, 0,
        nullptr, qkvb, 1, 0, 0, 3 * EMB,
        MROWS, 3 * EMB, EMB, 1.0f, 0);

    // ---- split/pad Q,K (row=token) and V^T (row=d) ----
    {
      const size_t total = (size_t)BH * KPAD * HDIM;
      split_qkv<<<(unsigned)((total + 255) / 256), 256, 0, stream>>>(qkvb, Qb, Kb, Vt);
    }

    // ---- scores: S = scale * Q @ K^T  (B operand = Kb, already [n=token,k=d]) ----
    gemm_bf16_wmma<<<dim3(cdiv_h(NTOK, 64), cdiv_h(NTOK, 128), BH), 256, 0, stream>>>(
        Qb, (long long)NPAD * HDIM, HDIM,
        Kb, (long long)NPAD * HDIM, HDIM,
        nullptr, nullptr, 0, 0,
        Sb, nullptr, 1, (long long)NPAD * KPAD, 0, KPAD,
        NTOK, NTOK, HDIM, attn_scale, 0);

    // ---- softmax -> P (bf16) ----
    softmax_rows<<<BH * NTOK, 224, 0, stream>>>(Sb, Pb);

    // ---- attention statistics (last block only) ----
    if (l == DEPTH - 1) {
      attn_back_mean<<<cdiv_h(BATCH * NTOK * NTOK, 256), 256, 0, stream>>>(Pb, out_ab);
      cls_attn_mean<<<cdiv_h(BATCH * NTOK, 256), 256, 0, stream>>>(Pb, cab);
      cls_patch_kernel<<<cdiv_h(BATCH * 196, 256), 256, 0, stream>>>(out_ab, cab, out_cp);
    }

    // ---- O = P @ V  (B operand = Vt [n=d, k=token]) -> merged o (bf16) ----
    gemm_bf16_wmma<<<dim3(cdiv_h(HDIM, 64), cdiv_h(NTOK, 128), BH), 256, 0, stream>>>(
        Pb, (long long)NPAD * KPAD, KPAD,
        Vt, (long long)HDIM * KPAD, KPAD,
        nullptr, nullptr, 0, 0,
        nullptr, ob, NHEAD, (long long)NTOK * EMB, HDIM, EMB,
        NTOK, HDIM, KPAD, 1.0f, 0);

    // ---- proj + residual:  xB = xA + o @ Wp + b ----
    gemm_bf16_wmma<<<dim3(cdiv_h(EMB, 64), cdiv_h(MROWS, 128), 1), 256, 0, stream>>>(
        ob, 0, EMB, wp, 0, EMB,
        proj_b + (size_t)l * EMB,
        xA, 0, EMB,
        xB, nullptr, 1, 0, 0, EMB,
        MROWS, EMB, EMB, 1.0f, 0);

    // ---- LN2 -> ybuf ----
    ln_rows<<<MROWS, 256, 0, stream>>>(xB, EMB, ln2_s + (size_t)l * EMB, ln2_b + (size_t)l * EMB,
                                       ybuf, nullptr, EMB);

    // ---- fc1 + GELU -> h1 (bf16) ----
    gemm_bf16_wmma<<<dim3(cdiv_h(DFF, 64), cdiv_h(MROWS, 128), 1), 256, 0, stream>>>(
        ybuf, 0, EMB, w1, 0, EMB,
        fc1_b + (size_t)l * DFF,
        nullptr, 0, 0,
        nullptr, h1, 1, 0, 0, DFF,
        MROWS, DFF, EMB, 1.0f, 1);

    // ---- fc2 + residual:  xA = xB + h1 @ W2 + b ----
    gemm_bf16_wmma<<<dim3(cdiv_h(EMB, 64), cdiv_h(MROWS, 128), 1), 256, 0, stream>>>(
        h1, 0, DFF, w2, 0, DFF,
        fc2_b + (size_t)l * EMB,
        xB, 0, EMB,
        xA, nullptr, 1, 0, 0, EMB,
        MROWS, EMB, DFF, 1.0f, 0);
  }

  // ---- final LN on cls rows -> d_out[0 : 32*768] (fp32) ----
  ln_rows<<<BATCH, 256, 0, stream>>>(xA, (long long)NTOK * EMB, lnf_s, lnf_b,
                                     nullptr, out_cls, EMB);
}
